// AttentionWithRotaryPositionalEmbedding_87875030876866
// MI455X (gfx1250) — matrix operand order repaired and checked by reading, not compiled, not run on hardware
//
#include <hip/hip_runtime.h>

typedef __attribute__((ext_vector_type(16))) __bf16 v16bf;
typedef __attribute__((ext_vector_type(8)))  __bf16 v8bf;
typedef __attribute__((ext_vector_type(8)))  float  v8f;
typedef __attribute__((ext_vector_type(4)))  int    v4i;

#define B_ 8
#define N_ 2048
#define C_ 512
#define H_ 8
#define D_ 64
#define F_ 1536

#if defined(__has_builtin)
#  if __has_builtin(__builtin_amdgcn_global_load_async_to_lds_b128)
#    define ASYNC_LDS 1
#  endif
#  if __has_builtin(__builtin_amdgcn_s_wait_asynccnt)
#    define HAVE_WAIT_ASYNC 1
#  endif
#endif

typedef __attribute__((address_space(1))) v4i* g4ptr;
typedef __attribute__((address_space(3))) v4i* l4ptr;

// 16-byte global -> LDS copy; async (ASYNCcnt-tracked) when available.
static __device__ inline void cp16(void* lds, const void* g) {
#ifdef ASYNC_LDS
  __builtin_amdgcn_global_load_async_to_lds_b128((g4ptr)(void*)g, (l4ptr)lds, 0, 0);
#else
  *(uint4*)lds = *(const uint4*)g;
#endif
}

static __device__ inline void wait_async_copies() {
#if defined(ASYNC_LDS) && defined(HAVE_WAIT_ASYNC)
  __builtin_amdgcn_s_wait_asynccnt(0);
#endif
}

static __device__ inline v8f wmma_bf16(v16bf a, v16bf b, v8f c) {
  // D = A(16x32 bf16) * B(32x16 bf16) + C(16x16 f32)
  return __builtin_amdgcn_wmma_f32_16x16x32_bf16(false, a, false, b, (short)0, c,
                                                 false, false);
}

static __device__ inline v16bf cat8(v8bf a, v8bf b) {
  return __builtin_shufflevector(a, b, 0, 1, 2, 3, 4, 5, 6, 7, 8, 9, 10, 11, 12,
                                 13, 14, 15);
}

// A-fragment (contiguous k order): per lane 16 contiguous elems at
// byte offset hi*32 of the 64-byte row.
static __device__ inline v16bf load_afrag(const __bf16* row, int hi) {
  v8bf a0 = *(const v8bf*)(row + hi * 16);
  v8bf a1 = *(const v8bf*)(row + hi * 16 + 8);
  return cat8(a0, a1);
}

static __device__ inline float rowmax16(float v) {
  #pragma unroll
  for (int m = 1; m < 16; m <<= 1) v = fmaxf(v, __shfl_xor(v, m, 32));
  return v;
}
static __device__ inline float rowsum16(float v) {
  #pragma unroll
  for (int m = 1; m < 16; m <<= 1) v += __shfl_xor(v, m, 32);
  return v;
}

// convert 8 consecutive f32 -> 8 bf16, one b128 LDS store
static __device__ inline void cvt8_store(__bf16* dst, const float* src) {
  float4 f0 = *(const float4*)src;
  float4 f1 = *(const float4*)(src + 4);
  __attribute__((aligned(16))) __bf16 t[8] = {
      (__bf16)f0.x, (__bf16)f0.y, (__bf16)f0.z, (__bf16)f0.w,
      (__bf16)f1.x, (__bf16)f1.y, (__bf16)f1.z, (__bf16)f1.w};
  *(uint4*)dst = *(const uint4*)t;
}

// ---------------------------------------------------------------------------
// Kernel 1: qkv = x @ W_qkv^T, fused RoPE(+scale on Q), bf16 outputs.
// 64(M) x 64(F) tile, 8 waves. Q,K -> [B][H][N][D]; V -> [B][H][D][N]
// (V transposed so P@V B-fragments are contiguous LDS reads in kernel 2).
// ---------------------------------------------------------------------------
__global__ void __launch_bounds__(256)
qkv_rope_kernel(const float* __restrict__ x, const float* __restrict__ Wqkv,
                const int* __restrict__ ncls_p,
                __bf16* __restrict__ Qs, __bf16* __restrict__ Ks,
                __bf16* __restrict__ Vs) {
  __shared__ __align__(16) __bf16 As[64][40];  // 80B rows: 16B aligned chunks
  __shared__ __align__(16) __bf16 Bs[64][40];
  const int tid  = threadIdx.x;
  const int lane = tid & 31;
  const int wid  = tid >> 5;
  const int wm   = wid >> 1;   // 0..3
  const int wn   = wid & 1;    // 0..1
  const int lo   = lane & 15;
  const int hi   = lane >> 4;
  const int srow = tid >> 2;        // staging row 0..63
  const int scol = (tid & 3) * 8;   // staging col 0,8,16,24
  const int rowBase = blockIdx.y * 64;  // over B*N
  const int colBase = blockIdx.x * 64;  // over 3C
  const int nc = *ncls_p;

  v8f acc0 = {};
  v8f acc1 = {};

  for (int k0 = 0; k0 < C_; k0 += 32) {
    cvt8_store(&As[srow][scol], &x[(size_t)(rowBase + srow) * C_ + k0 + scol]);
    cvt8_store(&Bs[srow][scol], &Wqkv[(size_t)(colBase + srow) * C_ + k0 + scol]);
    if (k0 + 32 < C_)
      __builtin_prefetch(&x[(size_t)(rowBase + srow) * C_ + k0 + 32 + scol], 0, 0);
    __syncthreads();

    const v16bf a = load_afrag(&As[wm * 16 + lo][0], hi);
    #pragma unroll
    for (int t = 0; t < 2; ++t) {
      const v16bf bfrag = *(const v16bf*)&Bs[wn * 32 + t * 16 + lo][hi * 16];
      if (t == 0) acc0 = wmma_bf16(a, bfrag, acc0);
      else        acc1 = wmma_bf16(a, bfrag, acc1);
    }
    __syncthreads();
  }

  // Epilogue: RoPE + scatter. C-layout: n = lo, m = r + 8*hi.
  const float scale = 0.125f;  // D^-0.5
  #pragma unroll
  for (int t = 0; t < 2; ++t) {
    v8f acc = t ? acc1 : acc0;
    const int f     = colBase + wn * 32 + t * 16 + lo;
    const int which = f / C_;  // 0=q 1=k 2=v
    const int cin   = f % C_;
    const int h     = cin >> 6;
    const int d     = cin & 63;
    const float inv_freq = __powf(10000.f, -(float)(2 * (d >> 1)) / (float)D_);
    #pragma unroll
    for (int r = 0; r < 8; ++r) {
      const int m   = rowBase + wm * 16 + r + 8 * hi;
      const int b   = m >> 11;       // / N_
      const int pos = m & (N_ - 1);
      float v   = acc[r];
      float prt = __shfl_xor(v, 1, 32);  // even/odd feature partner
      float outv = v;
      if (which < 2 && pos >= nc) {
        const float ang = (float)(pos - nc) * inv_freq;
        float sn, cs;
        __sincosf(ang, &sn, &cs);
        outv = ((d & 1) == 0) ? (v * cs - prt * sn)   // r_un
                              : (prt * sn + v * cs);  // r_ev
      }
      if (which == 0) {
        outv *= scale;
        Qs[(((size_t)(b * H_ + h)) * N_ + pos) * D_ + d] = (__bf16)outv;
      } else if (which == 1) {
        Ks[(((size_t)(b * H_ + h)) * N_ + pos) * D_ + d] = (__bf16)outv;
      } else {  // V transposed [D][N]
        Vs[(((size_t)(b * H_ + h)) * D_ + d) * N_ + pos] = (__bf16)outv;
      }
    }
  }
}

// ---------------------------------------------------------------------------
// Kernel 2: flash attention. 4 waves/block, each wave owns 16 query rows.
// Per 32-key chunk: async-stage K[32][64] and V^T[64][32] tiles to LDS
// (shared by all 4 waves), scores via 2x2 WMMA, online softmax, P through
// LDS (C-layout -> A-layout), o += P@V (4 WMMA). mask[query] cancels in
// softmax, so only mask[key] is applied.
// ---------------------------------------------------------------------------
__global__ void __launch_bounds__(128)
flash_attn_kernel(const float* __restrict__ mask,
                  const __bf16* __restrict__ Qs, const __bf16* __restrict__ Ks,
                  const __bf16* __restrict__ Vs, __bf16* __restrict__ AO) {
  __shared__ __align__(16) __bf16 Kt[32][72];     // [key][d]   144B rows
  __shared__ __align__(16) __bf16 Vt[64][40];     // [d][key]    80B rows
  __shared__ __align__(16) __bf16 Pt[4][16][40];  // per-wave P  80B rows
  const int tid  = threadIdx.x;
  const int lane = tid & 31;
  const int wid  = tid >> 5;
  const int lo   = lane & 15;
  const int hi   = lane >> 4;
  const int bh   = blockIdx.y;
  const int b    = bh >> 3;  // / H_
  const int h    = bh & 7;
  const int qm0  = blockIdx.x * 64 + wid * 16;

  // Q fragments once (A-layout; D=64 -> two 32-wide K-steps), b128 loads
  v16bf aq[2];
  {
    const __bf16* qrow = Qs + ((size_t)bh * N_ + (qm0 + lo)) * D_;
    #pragma unroll
    for (int kc = 0; kc < 2; ++kc) aq[kc] = load_afrag(qrow + kc * 32, hi);
  }

  float mrow[8], lsum[8];
  #pragma unroll
  for (int r = 0; r < 8; ++r) { mrow[r] = -1e30f; lsum[r] = 0.f; }
  v8f o[4] = {};

  for (int j0 = 0; j0 < N_; j0 += 32) {
    // ---- async-stage K (32x128B) and V^T (64x64B) chunks: 4 x b128/thread
    #pragma unroll
    for (int q = 0; q < 2; ++q) {
      const int ck = tid + q * 128;  // 0..255
      const int kr = ck >> 3, kc8 = (ck & 7) * 8;
      cp16(&Kt[kr][kc8], Ks + ((size_t)bh * N_ + j0 + kr) * D_ + kc8);
      const int vr = ck >> 2, vc8 = (ck & 3) * 8;
      cp16(&Vt[vr][vc8], Vs + ((size_t)bh * D_ + vr) * N_ + j0 + vc8);
    }
    wait_async_copies();
    __syncthreads();

    // ---- scores: s[jt] covers keys j0+jt*16..+15
    v8f s[2] = {};
    #pragma unroll
    for (int jt = 0; jt < 2; ++jt) {
      #pragma unroll
      for (int kc = 0; kc < 2; ++kc) {
        const v16bf bk = *(const v16bf*)&Kt[jt * 16 + lo][kc * 32 + hi * 16];
        s[jt] = wmma_bf16(aq[kc], bk, s[jt]);
      }
      const float mv = mask[b * N_ + j0 + jt * 16 + lo];  // key-side mask
      #pragma unroll
      for (int r = 0; r < 8; ++r) s[jt][r] += mv;
    }

    // ---- online softmax (row m = r + 8*hi; same mapping for s and o)
    float corr[8];
    #pragma unroll
    for (int r = 0; r < 8; ++r) {
      float cm = rowmax16(fmaxf(s[0][r], s[1][r]));
      const float nm = fmaxf(mrow[r], cm);
      corr[r] = __expf(mrow[r] - nm);
      mrow[r] = nm;
      const float p0 = __expf(s[0][r] - nm);
      const float p1 = __expf(s[1][r] - nm);
      s[0][r] = p0;
      s[1][r] = p1;
      lsum[r] = lsum[r] * corr[r] + rowsum16(p0 + p1);
    }
    #pragma unroll
    for (int t = 0; t < 4; ++t)
      #pragma unroll
      for (int r = 0; r < 8; ++r) o[t][r] *= corr[r];

    // ---- P: C-layout -> LDS -> A-layout
    #pragma unroll
    for (int jt = 0; jt < 2; ++jt)
      #pragma unroll
      for (int r = 0; r < 8; ++r)
        Pt[wid][r + 8 * hi][jt * 16 + lo] = (__bf16)s[jt][r];
    __syncthreads();
    const v16bf ap = load_afrag(&Pt[wid][lo][0], hi);

    // ---- o += P @ V   (B[k=key][n=d] from V^T tile: contiguous reads)
    #pragma unroll
    for (int t = 0; t < 4; ++t) {
      const v16bf bv = *(const v16bf*)&Vt[t * 16 + lo][hi * 16];
      o[t] = wmma_bf16(ap, bv, o[t]);
    }
    __syncthreads();
  }

  // ---- normalize and store attn-out as [B][N][H*D] bf16
  #pragma unroll
  for (int r = 0; r < 8; ++r) {
    const float inv = 1.f / lsum[r];
    const int npos = qm0 + r + 8 * hi;
    #pragma unroll
    for (int t = 0; t < 4; ++t)
      AO[((size_t)b * N_ + npos) * C_ + h * D_ + t * 16 + lo] =
          (__bf16)(o[t][r] * inv);
  }
}

// ---------------------------------------------------------------------------
// Kernel 3: out = AO @ W_proj^T + b_proj (fp32 out). A tile (already bf16)
// async-copied to LDS; B tile converted f32->bf16.
// ---------------------------------------------------------------------------
__global__ void __launch_bounds__(256)
proj_kernel(const __bf16* __restrict__ AO, const float* __restrict__ Wp,
            const float* __restrict__ bias, float* __restrict__ out) {
  __shared__ __align__(16) __bf16 As[64][40];
  __shared__ __align__(16) __bf16 Bs[64][40];
  const int tid  = threadIdx.x;
  const int lane = tid & 31;
  const int wid  = tid >> 5;
  const int wm   = wid >> 1;
  const int wn   = wid & 1;
  const int lo   = lane & 15;
  const int hi   = lane >> 4;
  const int srow = tid >> 2;
  const int scol = (tid & 3) * 8;
  const int rowBase = blockIdx.y * 64;
  const int colBase = blockIdx.x * 64;

  v8f acc0 = {};
  v8f acc1 = {};

  for (int k0 = 0; k0 < C_; k0 += 32) {
    cp16(&As[srow][scol], AO + (size_t)(rowBase + srow) * C_ + k0 + scol);
    cvt8_store(&Bs[srow][scol], &Wp[(size_t)(colBase + srow) * C_ + k0 + scol]);
    wait_async_copies();
    __syncthreads();

    const v16bf a = load_afrag(&As[wm * 16 + lo][0], hi);
    #pragma unroll
    for (int t = 0; t < 2; ++t) {
      const v16bf bfrag = *(const v16bf*)&Bs[wn * 32 + t * 16 + lo][hi * 16];
      if (t == 0) acc0 = wmma_bf16(a, bfrag, acc0);
      else        acc1 = wmma_bf16(a, bfrag, acc1);
    }
    __syncthreads();
  }

  #pragma unroll
  for (int t = 0; t < 2; ++t) {
    v8f acc = t ? acc1 : acc0;
    const int col = colBase + wn * 32 + t * 16 + lo;
    const float bv = bias[col];
    #pragma unroll
    for (int r = 0; r < 8; ++r) {
      const int m = rowBase + wm * 16 + r + 8 * hi;
      out[(size_t)m * C_ + col] = acc[r] + bv;
    }
  }
}

// ---------------------------------------------------------------------------
extern "C" void kernel_launch(void* const* d_in, const int* in_sizes, int n_in,
                              void* d_out, int out_size, void* d_ws, size_t ws_size,
                              hipStream_t stream) {
  const float* x    = (const float*)d_in[0];
  const float* mask = (const float*)d_in[1];
  const float* Wqkv = (const float*)d_in[2];
  const float* Wp   = (const float*)d_in[3];
  const float* bp   = (const float*)d_in[4];
  const int*   ncls = (const int*)d_in[5];
  float* out = (float*)d_out;

  __bf16* ws = (__bf16*)d_ws;
  const size_t QSZ = (size_t)B_ * H_ * N_ * D_;  // 8.39M elems each
  __bf16* Qs = ws;
  __bf16* Ks = ws + QSZ;
  __bf16* Vs = ws + 2 * QSZ;       // V stored transposed [B][H][D][N]
  __bf16* AO = ws + 3 * QSZ;       // 64 MB total bf16 workspace

  dim3 g1(F_ / 64, (B_ * N_) / 64);  // 24 x 256
  hipLaunchKernelGGL(qkv_rope_kernel, g1, dim3(256), 0, stream,
                     x, Wqkv, ncls, Qs, Ks, Vs);

  dim3 g2(N_ / 64, B_ * H_);  // 32 x 64
  hipLaunchKernelGGL(flash_attn_kernel, g2, dim3(128), 0, stream,
                     mask, Qs, Ks, Vs, AO);

  dim3 g3(C_ / 64, (B_ * N_) / 64);  // 8 x 256
  hipLaunchKernelGGL(proj_kernel, g3, dim3(256), 0, stream, AO, Wp, bp, out);
}
